// PINNDamageLocator_3865470566686
// MI455X (gfx1250) — matrix operand order repaired and checked by reading, not compiled, not run on hardware
//
#include <hip/hip_runtime.h>
#include <hip/hip_bf16.h>

typedef __attribute__((ext_vector_type(16))) _Float16 v16h;
typedef __attribute__((ext_vector_type(8)))  float    v8f;

union FragH { uint4 q[2]; v16h v; };

static __device__ __forceinline__ float warp_sum32(float v) {
#pragma unroll
  for (int m = 16; m >= 1; m >>= 1) v += __shfl_xor(v, m, 32);
  return v;
}

// ---------------------------------------------------------------------------
// Fold conv bias + eval-mode BN into per-channel scale/bias; cast weights to
// f16 laid out [Cout][Kpad] (K = Cin*3*3, zero padded).
// ---------------------------------------------------------------------------
template<int CIN, int COUT, int K, int KPAD>
__global__ __launch_bounds__(256)
void prep_conv_k(const float* __restrict__ w,  const float* __restrict__ cb,
                 const float* __restrict__ bs, const float* __restrict__ bb,
                 const float* __restrict__ bm, const float* __restrict__ bv,
                 _Float16* __restrict__ wp, float* __restrict__ sb)
{
  int i = blockIdx.x * blockDim.x + threadIdx.x;
  if (i < COUT * KPAD) {
    int oc = i / KPAD, k = i % KPAD;
    wp[i] = (k < K) ? (_Float16)w[oc * K + k] : (_Float16)0.f;
  }
  if (i < COUT) {
    float inv = bs[i] * rsqrtf(bv[i] + 1e-5f);
    sb[i]        = inv;
    sb[COUT + i] = (cb[i] - bm[i]) * inv + bb[i];
  }
}

// ---------------------------------------------------------------------------
// Implicit-GEMM conv (3x3, pad 1, stride (1,2)) on v_wmma_f32_16x16x32_f16.
// One wave = 16 output cols x (POOL ? 2 : 1) conv rows x all Cout tiles.
// POOL fuses the following 2x2 maxpool (row max in regs, col max via shfl).
// A (weights) staged block-wide in LDS; input halo slab staged per-wave in LDS.
// ---------------------------------------------------------------------------
template<int CIN, int COUT, int H, int W, int K, int KPAD, bool POOL, typename TIN>
__global__ __launch_bounds__(256)
void conv_wmma_k(const TIN* __restrict__ in, const _Float16* __restrict__ wp,
                 const float* __restrict__ sb, _Float16* __restrict__ out, int nimg)
{
  constexpr int WOUT    = W / 2;
  constexpr int TILES_W = WOUT / 16;
  constexpr int HITER   = POOL ? H / 2 : H;
  constexpr int ROWS    = POOL ? 2 : 1;
  constexpr int SROWS   = ROWS + 2;                 // conv rows + halo
  constexpr int KSTEPS  = KPAD / 32;
  constexpr int OCT     = COUT / 16;
  constexpr int SLAB    = CIN * SROWS * 34;         // 33 cols used + pad
  constexpr int SLABP   = (SLAB + 7) & ~7;

  extern __shared__ char smem_raw[];
  _Float16* wlds  = (_Float16*)smem_raw;            // [COUT][KPAD]
  _Float16* slab0 = wlds + COUT * KPAD;             // 8 per-wave slabs

  const int tid  = threadIdx.x;
  const int wave = tid >> 5;
  const int lane = tid & 31;
  const int half = lane >> 4;
  const int l15  = lane & 15;

  for (int i = tid; i < COUT * KPAD; i += 256) wlds[i] = wp[i];
  __syncthreads();

  const int total = nimg * HITER * TILES_W;
  const int t = blockIdx.x * 8 + wave;
  if (t >= total) return;                           // wave-uniform

  const int tw  = t % TILES_W;
  const int rr0 = t / TILES_W;
  const int hp  = rr0 % HITER;
  const int n   = rr0 / HITER;
  const int h0  = POOL ? 2 * hp : hp;
  const int wb  = tw * 16;
  const int cb_ = 2 * wb - 1;

  __builtin_prefetch(in + ((long)(n * CIN) * H + h0) * (long)W + (cb_ < 0 ? 0 : cb_), 0, 1);

  _Float16* slab = slab0 + wave * SLABP;
  for (int i = lane; i < SLAB; i += 32) {           // stage halo slab (zero pad)
    int c  = i % 34;
    int rr = (i / 34) % SROWS;
    int ic = i / (34 * SROWS);
    int gr = h0 - 1 + rr;
    int gc = cb_ + c;
    float v = 0.f;
    if (gr >= 0 && gr < H && gc >= 0 && gc < W)
      v = (float)in[((long)(n * CIN + ic) * H + gr) * (long)W + gc];
    slab[i] = (_Float16)v;
  }

  v8f acc[OCT][ROWS];
#pragma unroll
  for (int o = 0; o < OCT; ++o)
#pragma unroll
    for (int r = 0; r < ROWS; ++r) acc[o][r] = {};

#pragma unroll
  for (int s = 0; s < KSTEPS; ++s) {
    v16h bf[ROWS];
#pragma unroll
    for (int r = 0; r < ROWS; ++r) {
      // B layout: lane holds col N=l15, K = 32s + 16*half + j (16 contiguous K)
#pragma unroll
      for (int j = 0; j < 16; ++j) {
        int kk = 32 * s + 16 * half + j;
        _Float16 v = (_Float16)0.f;
        if (kk < K) {
          int ic   = kk / 9;
          int rem9 = kk - ic * 9;
          int kh   = rem9 / 3;
          int kw   = rem9 - kh * 3;
          v = slab[(ic * SROWS + (kh + r)) * 34 + 2 * l15 + kw];
        }
        bf[r][j] = v;
      }
    }
#pragma unroll
    for (int o = 0; o < OCT; ++o) {
      // A layout: lane holds row M=l15; K = 32s + 16*half + j (16 contiguous K)
      FragH af;
      const _Float16* wr = wlds + (o * 16 + l15) * KPAD + 32 * s + half * 16;
      af.q[0] = *(const uint4*)(wr);
      af.q[1] = *(const uint4*)(wr + 8);
#pragma unroll
      for (int r = 0; r < ROWS; ++r)
        acc[o][r] = __builtin_amdgcn_wmma_f32_16x16x32_f16(
            false, af.v, false, bf[r], (short)0, acc[o][r], false, false);
    }
  }

#pragma unroll
  for (int o = 0; o < OCT; ++o) {
#pragma unroll
    for (int v = 0; v < 8; ++v) {
      int oc = o * 16 + v + half * 8;               // D: M = v + 8*half, N = l15
      float scv = sb[oc], biv = sb[COUT + oc];
      if constexpr (POOL) {
        float a0 = fmaxf(0.f, acc[o][0][v] * scv + biv);
        float a1 = fmaxf(0.f, acc[o][1][v] * scv + biv);
        float m  = fmaxf(a0, a1);
        float mm = fmaxf(m, __shfl_xor(m, 1, 32));  // width-pair max
        if ((lane & 1) == 0) {
          int wq = tw * 8 + (l15 >> 1);
          out[(((long)n * COUT + oc) * (H / 2) + hp) * (WOUT / 2) + wq] = (_Float16)mm;
        }
      } else {
        float a0 = fmaxf(0.f, acc[o][0][v] * scv + biv);
        out[(((long)n * COUT + oc) * H + hp) * WOUT + wb + l15] = (_Float16)a0;
      }
    }
  }
}

// ---------------------------------------------------------------------------
// SE block pieces + global average pool
// ---------------------------------------------------------------------------
__global__ __launch_bounds__(256)
void se_mean_k(const _Float16* __restrict__ p, float* __restrict__ mean, int nc, int hw)
{
  int id = blockIdx.x * 8 + (threadIdx.x >> 5);
  if (id >= nc) return;
  int lane = threadIdx.x & 31;
  const _Float16* base = p + (long)id * hw;
  float s = 0.f;
  for (int i = lane; i < hw; i += 32) s += (float)base[i];
  s = warp_sum32(s);
  if (lane == 0) mean[id] = s / (float)hw;
}

__global__ __launch_bounds__(64)
void se_gate_k(const float* __restrict__ mean, const float* __restrict__ w1,
               const float* __restrict__ b1, const float* __restrict__ w2,
               const float* __restrict__ b2, float* __restrict__ gate, int C, int Hd)
{
  __shared__ float y[32];
  int n = blockIdx.x, t = threadIdx.x;
  if (t < Hd) {
    float s = b1[t];
    for (int c = 0; c < C; ++c) s += mean[n * C + c] * w1[t * C + c];
    y[t] = fmaxf(0.f, s);
  }
  __syncthreads();
  if (t < C) {
    float s = b2[t];
    for (int h = 0; h < Hd; ++h) s += y[h] * w2[t * Hd + h];
    gate[n * C + t] = 1.f / (1.f + expf(-s));
  }
}

__global__ __launch_bounds__(256)
void se_scale_k(_Float16* __restrict__ p, const float* __restrict__ gate, int hw, long total)
{
  long i = (long)blockIdx.x * blockDim.x + threadIdx.x;
  long stride = (long)gridDim.x * blockDim.x;
  for (; i < total; i += stride) {
    long nc = i / hw;
    p[i] = (_Float16)((float)p[i] * gate[nc]);
  }
}

__global__ __launch_bounds__(256)
void gap_k(const _Float16* __restrict__ h3, _Float16* __restrict__ h16, int nc, int hw)
{
  int id = blockIdx.x * 8 + (threadIdx.x >> 5);
  if (id >= nc) return;
  int lane = threadIdx.x & 31;
  const _Float16* base = h3 + (long)id * hw;
  float s = 0.f;
  for (int i = lane; i < hw; i += 32) s += (float)base[i];
  s = warp_sum32(s);
  if (lane == 0) h16[id] = (_Float16)(s / (float)hw);
}

// ---------------------------------------------------------------------------
// edge_fc: ef[288,64] = relu(h16[288,64] @ W[64,64]^T + b) via WMMA (4x18 tiles)
// ---------------------------------------------------------------------------
__global__ __launch_bounds__(256)
void edgefc_k(const float* __restrict__ wf32, const float* __restrict__ bias,
              const _Float16* __restrict__ h16, float* __restrict__ ef)
{
  extern __shared__ char smem_raw[];
  _Float16* wlds = (_Float16*)smem_raw;             // 64x64 f16
  int tid = threadIdx.x;
  for (int i = tid; i < 64 * 64; i += 256) wlds[i] = (_Float16)wf32[i];
  __syncthreads();

  int t = blockIdx.x * 8 + (tid >> 5);
  if (t >= 4 * 18) return;
  int lane = tid & 31, half = lane >> 4, l15 = lane & 15;
  int mt = t % 4, nt = t / 4;

  v8f acc = {};
#pragma unroll
  for (int s = 0; s < 2; ++s) {
    FragH af, bf;
    const _Float16* wr = wlds + (mt * 16 + l15) * 64 + 32 * s + half * 16;
    af.q[0] = *(const uint4*)wr;
    af.q[1] = *(const uint4*)(wr + 8);
    const _Float16* hr = h16 + ((long)nt * 16 + l15) * 64 + 32 * s + half * 16;
    bf.q[0] = *(const uint4*)hr;
    bf.q[1] = *(const uint4*)(hr + 8);
    acc = __builtin_amdgcn_wmma_f32_16x16x32_f16(false, af.v, false, bf.v,
                                                 (short)0, acc, false, false);
  }
#pragma unroll
  for (int v = 0; v < 8; ++v) {
    int oc = mt * 16 + v + half * 8;
    int sm = nt * 16 + l15;
    ef[(long)sm * 64 + oc] = fmaxf(0.f, acc[v] + bias[oc]);
  }
}

// ---------------------------------------------------------------------------
// Graph head: softplus edge weights, scatter to 12 nodes, e2n linear+relu,
// attention softmax, pooled heads + normalized edge attention. One wave per b.
// ---------------------------------------------------------------------------
__global__ __launch_bounds__(256)
void graph_k(const float* __restrict__ ef,  const float* __restrict__ edge_w,
             const float* __restrict__ e2nw, const float* __restrict__ e2nb,
             const float* __restrict__ a1w, const float* __restrict__ a1b,
             const float* __restrict__ a2w, const float* __restrict__ clsw,
             const float* __restrict__ clsb, const float* __restrict__ regw,
             const float* __restrict__ regb, float* __restrict__ out)
{
  extern __shared__ char smem_raw[];
  constexpr int PER = 2592;
  float* base = (float*)smem_raw + (threadIdx.x >> 5) * PER;
  float* spw = base;              // 36 (alloc 64)
  float* nin = base + 64;         // 12*64
  float* nd  = nin + 768;         // 12*128
  float* sc  = nd + 1536;         // 12 (alloc 16)
  float* aw  = sc + 16;           // 12 (alloc 16)
  float* gv  = aw + 16;           // 128
  float* ea  = gv + 128;          // 36
  int lane = threadIdx.x & 31;
  int b    = threadIdx.x >> 5;

  for (int e = lane; e < 36; e += 32) {
    float x = edge_w[e];
    spw[e] = fmaxf(x, 0.f) + log1pf(expf(-fabsf(x)));   // softplus
  }
  __syncthreads();

  for (int idx = lane; idx < 12 * 64; idx += 32) {      // node scatter-add
    int v = idx >> 6, d = idx & 63;
    float s = 0.f;
#pragma unroll
    for (int j = 0; j < 6; ++j) {
      int e = (v < 6) ? (v * 6 + j) : (j * 6 + (v - 6));
      s += spw[e] * ef[((long)b * 36 + e) * 64 + d];
    }
    nin[idx] = s;
  }
  __syncthreads();

  for (int idx = lane; idx < 12 * 128; idx += 32) {     // e2n + relu
    int v = idx >> 7, o = idx & 127;
    float s = e2nb[o];
    const float* wr = e2nw + o * 64;
    const float* xi = nin + v * 64;
    for (int d = 0; d < 64; ++d) s += wr[d] * xi[d];
    nd[idx] = fmaxf(0.f, s);
  }
  __syncthreads();

  if (lane < 12) {                                      // attention scores
    const float* xv = nd + lane * 128;
    float s = 0.f;
    for (int h = 0; h < 64; ++h) {
      float tt = a1b[h];
      const float* wr = a1w + h * 128;
      for (int d = 0; d < 128; ++d) tt += wr[d] * xv[d];
      s += a2w[h] * tanhf(tt);
    }
    sc[lane] = s;
  }
  __syncthreads();

  if (lane < 12) {                                      // softmax over nodes
    float m = -1e30f;
    for (int v = 0; v < 12; ++v) m = fmaxf(m, sc[v]);
    float den = 0.f;
    for (int v = 0; v < 12; ++v) den += expf(sc[v] - m);
    aw[lane] = expf(sc[lane] - m) / den;
  }
  __syncthreads();

  for (int o = lane; o < 128; o += 32) {                // pooled graph vec
    float s = 0.f;
    for (int v = 0; v < 12; ++v) s += nd[v * 128 + o] * aw[v];
    gv[o] = s;
  }
  for (int e = lane; e < 36; e += 32) ea[e] = aw[e / 6] * aw[6 + e % 6];
  __syncthreads();

  if (lane < 3) {                                       // reg head
    float s = regb[lane];
    for (int d = 0; d < 128; ++d) s += regw[lane * 128 + d] * gv[d];
    out[b * 3 + lane] = s;
  }
  if (lane < 9) {                                       // cls head
    float s = clsb[lane];
    for (int d = 0; d < 128; ++d) s += clsw[lane * 128 + d] * gv[d];
    out[24 + b * 9 + lane] = s;
  }
  float tot = 0.f;
  for (int e = 0; e < 36; ++e) tot += ea[e];
  float inv = 1.f / (tot + 1e-8f);
  for (int e = lane; e < 36; e += 32) out[96 + b * 36 + e] = ea[e] * inv;
}

// ---------------------------------------------------------------------------
extern "C" void kernel_launch(void* const* d_in, const int* in_sizes, int n_in,
                              void* d_out, int out_size, void* d_ws, size_t ws_size,
                              hipStream_t stream) {
  const float* x       = (const float*)d_in[0];
  const float* c1w = (const float*)d_in[1];  const float* c1b = (const float*)d_in[2];
  const float* b1s = (const float*)d_in[3];  const float* b1b = (const float*)d_in[4];
  const float* b1m = (const float*)d_in[5];  const float* b1v = (const float*)d_in[6];
  const float* s1w1 = (const float*)d_in[7]; const float* s1b1 = (const float*)d_in[8];
  const float* s1w2 = (const float*)d_in[9]; const float* s1b2 = (const float*)d_in[10];
  const float* c2w = (const float*)d_in[11]; const float* c2b = (const float*)d_in[12];
  const float* b2s = (const float*)d_in[13]; const float* b2b = (const float*)d_in[14];
  const float* b2m = (const float*)d_in[15]; const float* b2v = (const float*)d_in[16];
  const float* s2w1 = (const float*)d_in[17]; const float* s2b1 = (const float*)d_in[18];
  const float* s2w2 = (const float*)d_in[19]; const float* s2b2 = (const float*)d_in[20];
  const float* c3w = (const float*)d_in[21]; const float* c3b = (const float*)d_in[22];
  const float* b3s = (const float*)d_in[23]; const float* b3b = (const float*)d_in[24];
  const float* b3m = (const float*)d_in[25]; const float* b3v = (const float*)d_in[26];
  const float* efw = (const float*)d_in[27]; const float* efb = (const float*)d_in[28];
  const float* edw = (const float*)d_in[29];
  const float* e2nw = (const float*)d_in[30]; const float* e2nb = (const float*)d_in[31];
  const float* a1w = (const float*)d_in[32]; const float* a1b = (const float*)d_in[33];
  const float* a2w = (const float*)d_in[34];
  const float* clw = (const float*)d_in[35]; const float* clb = (const float*)d_in[36];
  const float* rgw = (const float*)d_in[37]; const float* rgb = (const float*)d_in[38];

  const int NI = 288;                       // B*E
  char* ws = (char*)d_ws;
  auto alloc = [&](size_t bytes) { char* p = ws; ws += (bytes + 255) & ~(size_t)255; return p; };

  _Float16* p1  = (_Float16*)alloc((size_t)NI * 16 * 16 * 512 * 2);  // pooled SE1 input
  _Float16* p2  = (_Float16*)alloc((size_t)NI * 32 * 8 * 128 * 2);
  _Float16* h3  = (_Float16*)alloc((size_t)NI * 64 * 8 * 64 * 2);
  _Float16* h16 = (_Float16*)alloc((size_t)NI * 64 * 2);
  _Float16* wp1 = (_Float16*)alloc(16 * 64 * 2);   float* sb1 = (float*)alloc(32 * 4);
  _Float16* wp2 = (_Float16*)alloc(32 * 160 * 2);  float* sb2 = (float*)alloc(64 * 4);
  _Float16* wp3 = (_Float16*)alloc(64 * 288 * 2);  float* sb3 = (float*)alloc(128 * 4);
  float* mean1 = (float*)alloc((size_t)NI * 16 * 4);
  float* gate1 = (float*)alloc((size_t)NI * 16 * 4);
  float* mean2 = (float*)alloc((size_t)NI * 32 * 4);
  float* gate2 = (float*)alloc((size_t)NI * 32 * 4);
  float* ef    = (float*)alloc((size_t)NI * 64 * 4);

  // weight prep
  prep_conv_k<4, 16, 36, 64><<<(16 * 64 + 255) / 256, 256, 0, stream>>>(c1w, c1b, b1s, b1b, b1m, b1v, wp1, sb1);
  prep_conv_k<16, 32, 144, 160><<<(32 * 160 + 255) / 256, 256, 0, stream>>>(c2w, c2b, b2s, b2b, b2m, b2v, wp2, sb2);
  prep_conv_k<32, 64, 288, 288><<<(64 * 288 + 255) / 256, 256, 0, stream>>>(c3w, c3b, b3s, b3b, b3m, b3v, wp3, sb3);

  // conv1 + bn + relu + maxpool (fused), f32 input
  {
    int total = NI * 16 * 64;                               // n * hp * wtiles
    size_t sh = (size_t)(16 * 64 + 8 * (((4 * 4 * 34) + 7) & ~7)) * 2;
    conv_wmma_k<4, 16, 32, 2048, 36, 64, true, float>
        <<<(total + 7) / 8, 256, sh, stream>>>(x, wp1, sb1, p1, NI);
  }
  // SE1
  se_mean_k<<<(NI * 16 + 7) / 8, 256, 0, stream>>>(p1, mean1, NI * 16, 16 * 512);
  se_gate_k<<<NI, 64, 0, stream>>>(mean1, s1w1, s1b1, s1w2, s1b2, gate1, 16, 4);
  {
    long total = (long)NI * 16 * 16 * 512;
    se_scale_k<<<(int)((total + 2047) / 2048), 256, 0, stream>>>(p1, gate1, 16 * 512, total);
  }
  // conv2 + bn + relu + maxpool (fused)
  {
    int total = NI * 8 * 16;
    size_t sh = (size_t)(32 * 160 + 8 * (((16 * 4 * 34) + 7) & ~7)) * 2;
    conv_wmma_k<16, 32, 16, 512, 144, 160, true, _Float16>
        <<<(total + 7) / 8, 256, sh, stream>>>(p1, wp2, sb2, p2, NI);
  }
  // SE2
  se_mean_k<<<(NI * 32 + 7) / 8, 256, 0, stream>>>(p2, mean2, NI * 32, 8 * 128);
  se_gate_k<<<NI, 64, 0, stream>>>(mean2, s2w1, s2b1, s2w2, s2b2, gate2, 32, 8);
  {
    long total = (long)NI * 32 * 8 * 128;
    se_scale_k<<<(int)((total + 2047) / 2048), 256, 0, stream>>>(p2, gate2, 8 * 128, total);
  }
  // conv3 + bn + relu (no pool)
  {
    int total = NI * 8 * 4;
    size_t sh = (size_t)(64 * 288 + 8 * (((32 * 3 * 34) + 7) & ~7)) * 2;
    conv_wmma_k<32, 64, 8, 128, 288, 288, false, _Float16>
        <<<(total + 7) / 8, 256, sh, stream>>>(p2, wp3, sb3, h3, NI);
  }
  // global average pool -> [288,64] f16
  gap_k<<<(NI * 64 + 7) / 8, 256, 0, stream>>>(h3, h16, NI * 64, 8 * 64);
  // edge_fc via WMMA -> ef [288,64] f32
  edgefc_k<<<9, 256, 64 * 64 * 2, stream>>>(efw, efb, h16, ef);
  // graph head -> d_out (reg[24] | cls[72] | ea[288])
  graph_k<<<1, 256, 8 * 2592 * sizeof(float), stream>>>(
      ef, edw, e2nw, e2nb, a1w, a1b, a2w, clw, clb, rgw, rgb, (float*)d_out);
}
